// pew_LSTM_90434831385329
// MI455X (gfx1250) — 16-bit and 8-bit matrix operand order repaired and checked by reading, compiled, not run on hardware
//
#include <hip/hip_runtime.h>

#define BB 128
#define TT 1024
#define II 64
#define HH 256
#define WEA 32

typedef __attribute__((ext_vector_type(16))) _Float16 v16h;
typedef __attribute__((ext_vector_type(8)))  float    v8f;
typedef __attribute__((ext_vector_type(16))) int      v16i;

static constexpr long long PRE_FLOATS = 5LL * BB * TT * HH;
static constexpr long long OFF_F16    = PRE_FLOATS * 4;
static constexpr int F16_WE  = 0;
static constexpr int F16_WIX = 4096;
static constexpr int F16_WGX = 12288;
static constexpr int F16_WFX = 20480;
static constexpr int F16_WOX = 28672;
static constexpr int F16_WD  = 36864;
static constexpr int F16_WW  = 45056;
static constexpr int F16_WM  = 53248;
static constexpr int F16_WIE = 61440;
static constexpr int F16_WFE = 94208;
static constexpr int F16_WOE = 126976;
static constexpr int F16_TOTAL = 159744;
static constexpr long long OFF_FP8 = OFF_F16 + (long long)F16_TOTAL * 4;
static constexpr int FP8_WT  = 0;
static constexpr int FP8_WIH = 16384;
static constexpr int FP8_WGH = 32768;
static constexpr int FP8_WFO = 49152;
static constexpr int FP8_WOH = 65536;

__device__ __forceinline__ v8f wmma16(v16h a, v16h b, v8f c) {
  return __builtin_amdgcn_wmma_f32_16x16x32_f16(false, a, false, b, (short)0, c, false, false);
}
__device__ __forceinline__ v8f wmma8(v16i a, v16i b, v8f c) {
  return __builtin_amdgcn_wmma_f32_16x16x128_fp8_fp8(a, b, (short)0, c, false, false);
}

__device__ __forceinline__ unsigned f32_to_fp8_e4m3(float x) {
  union { float f; unsigned u; } cv; cv.f = x;
  unsigned u = cv.u;
  unsigned s = (u >> 24) & 0x80u;
  int e = (int)((u >> 23) & 0xff);
  unsigned m = u & 0x7fffffu;
  if (e == 0xff) return s | 0x7e;
  int e8 = e - 120;
  if (e8 > 15) return s | 0x7e;
  if (e8 <= 0) {
    int shift = 21 - e8;
    if (shift > 24) return s;
    unsigned full = 0x800000u | m;
    unsigned d = (full >> shift) + ((full >> (shift - 1)) & 1u);
    return s | (d & 0x7fu);
  }
  unsigned v = ((unsigned)e8 << 3) | (m >> 20);
  v += (m >> 19) & 1u;
  if (v > 0x7eu) v = 0x7eu;
  return s | v;
}
__device__ __forceinline__ unsigned pk_fp8(float a, float b) {
#if __has_builtin(__builtin_amdgcn_cvt_pk_fp8_f32)
  return (unsigned)__builtin_amdgcn_cvt_pk_fp8_f32(a, b, 0, false) & 0xffffu;
#else
  return f32_to_fp8_e4m3(a) | (f32_to_fp8_e4m3(b) << 8);
#endif
}

__device__ __forceinline__ v16h ldsA_f32(const float* S, int stride, int row0, int lane, int kbase) {
  int M = lane & 15, hi = (lane >= 16) ? 8 : 0;
  v16h a;
#pragma unroll
  for (int v = 0; v < 8; ++v) {
    int k = kbase + ((v < 4) ? 0 : 16) + hi + (v & 3) * 2;
    const float* p = S + (row0 + M) * stride + k;
    a[2 * v]     = (_Float16)p[0];
    a[2 * v + 1] = (_Float16)p[1];
  }
  return a;
}
__device__ __forceinline__ v16h ldsA_f16(const _Float16* S, int lane, int kt) {
  int M = lane & 15, hi = (lane >= 16) ? 8 : 0;
  v16h a;
#pragma unroll
  for (int v = 0; v < 8; ++v) {
    int k = kt * 32 + ((v < 4) ? 0 : 16) + hi + (v & 3) * 2;
    a[2 * v]     = S[M * HH + k];
    a[2 * v + 1] = S[M * HH + k + 1];
  }
  return a;
}
__device__ __forceinline__ v16h ldB16(const unsigned* area, int ktiles, int nt, int kt, int lane) {
  return *(const v16h*)(area + (((nt * ktiles) + kt) * 32 + lane) * 8);
}
__device__ __forceinline__ v16i ldA8(const unsigned char* S, int lane, int kc) {
  int M = lane & 15, hi = (lane >= 16) ? 8 : 0;
  v16i a;
#pragma unroll
  for (int p = 0; p < 8; ++p) {
    int v = 2 * p;
    int k = kc * 128 + (v >> 3) * 64 + ((v >> 2) & 1) * 32 + ((v >> 1) & 1) * 16 + hi;
    uint2 d = *(const uint2*)(S + M * HH + k);
    a[v]     = (int)d.x;
    a[v + 1] = (int)d.y;
  }
  return a;
}
__device__ __forceinline__ v16i ldB8(const unsigned* gw, int m, int nt, int kc, int lane) {
  return *(const v16i*)(gw + (((m * 16 + nt) * 2 + kc) * 32 + lane) * 16);
}
__device__ __forceinline__ v8f csplat(const float* bias, int nt, int lane) {
  float s = bias[nt * 16 + (lane & 15)];
  v8f c;
#pragma unroll
  for (int v = 0; v < 8; ++v) c[v] = s;
  return c;
}
__device__ __forceinline__ void storePre(float* pre, int btile, int t, int m, int nt, int lane, v8f c) {
  float* dst = pre + ((((long long)(btile * TT + t) * 5 + m) * 16 + nt) * 32 + lane) * 8;
  *(v8f*)dst = c;
}
__device__ __forceinline__ float sigm(float x) {
  return __builtin_amdgcn_rcpf(1.f + __expf(-x));
}
__device__ __forceinline__ float tanh_f(float x) {
  float xx = fminf(15.f, fmaxf(-15.f, x));
  float e = __expf(2.f * xx);
  return (e - 1.f) * __builtin_amdgcn_rcpf(e + 1.f);
}

__global__ void k_cvt_f16(const float* __restrict__ W, unsigned* __restrict__ out, int K) {
  int idx = blockIdx.x * blockDim.x + threadIdx.x;
  int ktiles = K >> 5;
  int total = 16 * ktiles * 256;
  if (idx >= total) return;
  int v = idx & 7;
  int lane = (idx >> 3) & 31;
  int kt = (idx >> 8) % ktiles;
  int nt = (idx >> 8) / ktiles;
  int n = nt * 16 + (lane & 15);
  int k = kt * 32 + ((v < 4) ? 0 : 16) + ((lane >= 16) ? 8 : 0) + (v & 3) * 2;
  union { _Float16 h[2]; unsigned u; } p;
  p.h[0] = (_Float16)W[n * K + k];
  p.h[1] = (_Float16)W[n * K + k + 1];
  out[idx] = p.u;
}
__global__ void k_cvt_fp8(const float* __restrict__ W, unsigned* __restrict__ out) {
  int idx = blockIdx.x * blockDim.x + threadIdx.x;
  if (idx >= 16384) return;
  int v = idx & 15;
  int lane = (idx >> 4) & 31;
  int kc = (idx >> 9) & 1;
  int nt = idx >> 10;
  int n = nt * 16 + (lane & 15);
  int kb = kc * 128 + (v >> 1) * 16 + ((lane >= 16) ? 8 : 0) + (v & 1) * 4;
  const float* w = W + n * HH + kb;
  unsigned lo = pk_fp8(w[0], w[1]);
  unsigned hi = pk_fp8(w[2], w[3]);
  out[idx] = lo | (hi << 16);
}

__global__ __launch_bounds__(256) void k_precompute(
    const float* __restrict__ x, const float* __restrict__ xw,
    const float* __restrict__ b_i, const float* __restrict__ b_f,
    const float* __restrict__ b_o, const float* __restrict__ b_g,
    const float* __restrict__ b_e,
    const unsigned* __restrict__ f16a, float* __restrict__ pre) {
  __shared__ float xs[48 * II];
  __shared__ float wsh[16 * WEA];
  __shared__ _Float16 eS[16 * HH];
  const int t = blockIdx.x, btile = blockIdx.y;
  const int tid = threadIdx.x, lane = tid & 31, wave = tid >> 5;
  const int hi8 = (lane >= 16) ? 8 : 0;

  for (int i = tid; i < 48 * II; i += 256) {
    int r = i >> 6, k = i & 63;
    int b = btile * 16 - 30 + r;
    xs[i] = (b >= 0 && b < BB) ? x[((long long)b * TT + t) * II + k] : 0.f;
  }
  for (int i = tid; i < 16 * WEA; i += 256) {
    int r = i >> 5, k = i & 31;
    wsh[i] = xw[((long long)(btile * 16 + r) * TT + t) * WEA + k];
  }
  __syncthreads();

  v16h aW = ldsA_f32(wsh, WEA, 0, lane, 0);
#pragma unroll
  for (int i = 0; i < 2; ++i) {
    int nt = 2 * wave + i;
    v8f c = csplat(b_e, nt, lane);
    c = wmma16(aW, ldB16(f16a + F16_WE, 1, nt, 0, lane), c);
#pragma unroll
    for (int v = 0; v < 8; ++v)
      eS[(hi8 + v) * HH + nt * 16 + (lane & 15)] = (_Float16)sigm(c[v]);
  }
  __syncthreads();

  v16h xA0 = ldsA_f32(xs, II, 30, lane, 0), xA1 = ldsA_f32(xs, II, 30, lane, 32);
  v16h dA0 = ldsA_f32(xs, II, 29, lane, 0), dA1 = ldsA_f32(xs, II, 29, lane, 32);
  v16h wA0 = ldsA_f32(xs, II, 23, lane, 0), wA1 = ldsA_f32(xs, II, 23, lane, 32);
  v16h mA0 = ldsA_f32(xs, II, 0,  lane, 0), mA1 = ldsA_f32(xs, II, 0,  lane, 32);
  v16h eA[8];
#pragma unroll
  for (int kt = 0; kt < 8; ++kt) eA[kt] = ldsA_f16(eS, lane, kt);

#pragma unroll
  for (int i = 0; i < 2; ++i) {
    int nt = 2 * wave + i;
    v8f c = csplat(b_e, nt, lane);
    c = wmma16(dA0, ldB16(f16a + F16_WD, 2, nt, 0, lane), c);
    c = wmma16(dA1, ldB16(f16a + F16_WD, 2, nt, 1, lane), c);
    c = wmma16(wA0, ldB16(f16a + F16_WW, 2, nt, 0, lane), c);
    c = wmma16(wA1, ldB16(f16a + F16_WW, 2, nt, 1, lane), c);
    c = wmma16(mA0, ldB16(f16a + F16_WM, 2, nt, 0, lane), c);
    c = wmma16(mA1, ldB16(f16a + F16_WM, 2, nt, 1, lane), c);
    storePre(pre, btile, t, 0, nt, lane, c);
    c = csplat(b_i, nt, lane);
    c = wmma16(xA0, ldB16(f16a + F16_WIX, 2, nt, 0, lane), c);
    c = wmma16(xA1, ldB16(f16a + F16_WIX, 2, nt, 1, lane), c);
#pragma unroll
    for (int kt = 0; kt < 8; ++kt) c = wmma16(eA[kt], ldB16(f16a + F16_WIE, 8, nt, kt, lane), c);
    storePre(pre, btile, t, 1, nt, lane, c);
    c = csplat(b_g, nt, lane);
    c = wmma16(xA0, ldB16(f16a + F16_WGX, 2, nt, 0, lane), c);
    c = wmma16(xA1, ldB16(f16a + F16_WGX, 2, nt, 1, lane), c);
    storePre(pre, btile, t, 2, nt, lane, c);
    c = csplat(b_f, nt, lane);
    c = wmma16(xA0, ldB16(f16a + F16_WFX, 2, nt, 0, lane), c);
    c = wmma16(xA1, ldB16(f16a + F16_WFX, 2, nt, 1, lane), c);
#pragma unroll
    for (int kt = 0; kt < 8; ++kt) c = wmma16(eA[kt], ldB16(f16a + F16_WFE, 8, nt, kt, lane), c);
    storePre(pre, btile, t, 3, nt, lane, c);
    c = csplat(b_o, nt, lane);
    c = wmma16(xA0, ldB16(f16a + F16_WOX, 2, nt, 0, lane), c);
    c = wmma16(xA1, ldB16(f16a + F16_WOX, 2, nt, 1, lane), c);
#pragma unroll
    for (int kt = 0; kt < 8; ++kt) c = wmma16(eA[kt], ldB16(f16a + F16_WOE, 8, nt, kt, lane), c);
    storePre(pre, btile, t, 4, nt, lane, c);
  }
}

__global__ __launch_bounds__(256) void k_recurrent(
    const float* __restrict__ pre, const unsigned* __restrict__ fp8a,
    float* __restrict__ out_h, float* __restrict__ out_c) {
  extern __shared__ unsigned char smem[];
  unsigned* gw = (unsigned*)smem;
  unsigned char* hA  = smem + 262144;
  unsigned char* hoA = smem + 262144 + 4096;
  const int btile = blockIdx.x;
  const int tid = threadIdx.x, lane = tid & 31, wave = tid >> 5;
  const int hi8 = (lane >= 16) ? 8 : 0;

  {
    const uint4* src = (const uint4*)(fp8a + FP8_WIH);
    uint4* dst = (uint4*)gw;
    for (int i = tid; i < 16384; i += 256) dst[i] = src[i];
  }
  { uint4 z; z.x = z.y = z.z = z.w = 0u; ((uint4*)hA)[tid] = z; }

  v16i wt[2][2];
#pragma unroll
  for (int i = 0; i < 2; ++i)
#pragma unroll
    for (int kc = 0; kc < 2; ++kc)
      wt[i][kc] = *(const v16i*)(fp8a + FP8_WT + (((2 * wave + i) * 2 + kc) * 32 + lane) * 16);

  v8f cst[2];
#pragma unroll
  for (int v = 0; v < 8; ++v) { cst[0][v] = 0.f; cst[1][v] = 0.f; }
  __syncthreads();

  for (int t = 0; t < TT; ++t) {
    const long long pbase = (long long)(btile * TT + t) * (5 * 16 * 32 * 8);
    if (t + 1 < TT)
      __builtin_prefetch((const void*)(pre + pbase + 5 * 16 * 32 * 8 + wave * 2560), 0, 1);

    v16i a0 = ldA8(hA, lane, 0);
    v16i a1 = ldA8(hA, lane, 1);
#pragma unroll
    for (int i = 0; i < 2; ++i) {
      int nt = 2 * wave + i;
      v8f acc = *(const v8f*)(pre + pbase + ((0 * 16 + nt) * 32 + lane) * 8);
      acc = wmma8(a0, wt[i][0], acc);
      acc = wmma8(a1, wt[i][1], acc);
      int n = nt * 16 + (lane & 15);
#pragma unroll
      for (int p = 0; p < 4; ++p) {
        unsigned pk = pk_fp8(sigm(acc[2 * p]), sigm(acc[2 * p + 1]));
        hoA[(hi8 + 2 * p) * HH + n]     = (unsigned char)(pk & 0xff);
        hoA[(hi8 + 2 * p + 1) * HH + n] = (unsigned char)(pk >> 8);
      }
    }
    __syncthreads();

    a0 = ldA8(hoA, lane, 0);
    a1 = ldA8(hoA, lane, 1);
#pragma unroll
    for (int i = 0; i < 2; ++i) {
      int nt = 2 * wave + i;
      v8f g4[4];
#pragma unroll
      for (int m = 0; m < 4; ++m) {
        v8f acc = *(const v8f*)(pre + pbase + (((m + 1) * 16 + nt) * 32 + lane) * 8);
        acc = wmma8(a0, ldB8(gw, m, nt, 0, lane), acc);
        acc = wmma8(a1, ldB8(gw, m, nt, 1, lane), acc);
        g4[m] = acc;
      }
      int n = nt * 16 + (lane & 15);
      float hv[8];
#pragma unroll
      for (int v = 0; v < 8; ++v) {
        float ig = sigm(g4[0][v]);
        float gg = tanh_f(g4[1][v]);
        float fg = sigm(g4[2][v]);
        float og = sigm(g4[3][v]);
        float cn = fg * cst[i][v] + ig * gg;
        cst[i][v] = cn;
        float hn = og * tanh_f(cn);
        hv[v] = hn;
        int b = btile * 16 + hi8 + v;
        long long oi = ((long long)b * TT + t) * HH + n;
        out_h[oi] = hn;
        out_c[oi] = cn;
      }
#pragma unroll
      for (int p = 0; p < 4; ++p) {
        unsigned pk = pk_fp8(hv[2 * p], hv[2 * p + 1]);
        hA[(hi8 + 2 * p) * HH + n]     = (unsigned char)(pk & 0xff);
        hA[(hi8 + 2 * p + 1) * HH + n] = (unsigned char)(pk >> 8);
      }
    }
    __syncthreads();
  }
}

extern "C" void kernel_launch(void* const* d_in, const int* in_sizes, int n_in,
                              void* d_out, int out_size, void* d_ws, size_t ws_size,
                              hipStream_t stream) {
  (void)in_sizes; (void)n_in; (void)out_size; (void)ws_size;
  const float* x    = (const float*)d_in[0];
  const float* xw   = (const float*)d_in[1];
  const float* w_ix = (const float*)d_in[2];
  const float* w_ih = (const float*)d_in[3];
  const float* w_ie = (const float*)d_in[4];
  const float* b_i  = (const float*)d_in[5];
  const float* w_fx = (const float*)d_in[6];
  const float* w_fo = (const float*)d_in[7];
  const float* w_fe = (const float*)d_in[8];
  const float* b_f  = (const float*)d_in[9];
  const float* w_ox = (const float*)d_in[10];
  const float* w_oh = (const float*)d_in[11];
  const float* w_oe = (const float*)d_in[12];
  const float* b_o  = (const float*)d_in[13];
  const float* w_gx = (const float*)d_in[14];
  const float* w_gh = (const float*)d_in[15];
  const float* b_g  = (const float*)d_in[16];
  const float* w_d  = (const float*)d_in[17];
  const float* w_w  = (const float*)d_in[18];
  const float* w_m  = (const float*)d_in[19];
  const float* w_t  = (const float*)d_in[20];
  const float* w_e  = (const float*)d_in[21];
  const float* b_e  = (const float*)d_in[22];

  float*    pre  = (float*)d_ws;
  unsigned* f16a = (unsigned*)((char*)d_ws + OFF_F16);
  unsigned* fp8a = (unsigned*)((char*)d_ws + OFF_FP8);

  k_cvt_f16<<<16,  256, 0, stream>>>(w_e,  f16a + F16_WE,  32);
  k_cvt_f16<<<32,  256, 0, stream>>>(w_ix, f16a + F16_WIX, 64);
  k_cvt_f16<<<32,  256, 0, stream>>>(w_gx, f16a + F16_WGX, 64);
  k_cvt_f16<<<32,  256, 0, stream>>>(w_fx, f16a + F16_WFX, 64);
  k_cvt_f16<<<32,  256, 0, stream>>>(w_ox, f16a + F16_WOX, 64);
  k_cvt_f16<<<32,  256, 0, stream>>>(w_d,  f16a + F16_WD,  64);
  k_cvt_f16<<<32,  256, 0, stream>>>(w_w,  f16a + F16_WW,  64);
  k_cvt_f16<<<32,  256, 0, stream>>>(w_m,  f16a + F16_WM,  64);
  k_cvt_f16<<<128, 256, 0, stream>>>(w_ie, f16a + F16_WIE, 256);
  k_cvt_f16<<<128, 256, 0, stream>>>(w_fe, f16a + F16_WFE, 256);
  k_cvt_f16<<<128, 256, 0, stream>>>(w_oe, f16a + F16_WOE, 256);
  k_cvt_fp8<<<64,  256, 0, stream>>>(w_t,  fp8a + FP8_WT);
  k_cvt_fp8<<<64,  256, 0, stream>>>(w_ih, fp8a + FP8_WIH);
  k_cvt_fp8<<<64,  256, 0, stream>>>(w_gh, fp8a + FP8_WGH);
  k_cvt_fp8<<<64,  256, 0, stream>>>(w_fo, fp8a + FP8_WFO);
  k_cvt_fp8<<<64,  256, 0, stream>>>(w_oh, fp8a + FP8_WOH);

  k_precompute<<<dim3(TT, BB / 16), 256, 0, stream>>>(x, xw, b_i, b_f, b_o, b_g, b_e, f16a, pre);

  float* out_h = (float*)d_out;
  float* out_c = out_h + (long long)BB * TT * HH;
  k_recurrent<<<BB / 16, 256, 270336, stream>>>(pre, fp8a, out_h, out_c);
}
